// PointRendRefiner_87522843558300
// MI455X (gfx1250) — matrix operand order repaired and checked by reading, not compiled, not run on hardware
//
#include <hip/hip_runtime.h>
#include <hip/hip_bf16.h>

// ---------------------------------------------------------------------------
// PointRend refiner for MI455X (gfx1250, wave32).
//   B=8, hid=[8,16,512,512], fgr=[8,3,128,128], pha=[8,1,128,128],
//   mlp_w=[4,20], mlp_b=[4].  Output: fgr_out[8,3,512,512] ++ pha_out[8,1,512,512]
// Two iterations (128->256->512): upsample2x(align_corners) -> uncertainty
// -> exact radix top-K -> point-sample + WMMA MLP + scatter (fused).
// CDNA5 paths: v_wmma_f32_16x16x32_f16 (MLP), TDM tensor_load_to_lds +
// s_wait_tensorcnt (weight staging), global_prefetch_b8 (histogram scan).
// ---------------------------------------------------------------------------

#define NB 8          // batch
#define HSZ 512       // hid / final spatial size

typedef __attribute__((ext_vector_type(16))) _Float16 v16h;
typedef __attribute__((ext_vector_type(8)))  float    v8f;
typedef unsigned int u32x4 __attribute__((ext_vector_type(4)));
typedef int          i32x4 __attribute__((ext_vector_type(4)));
typedef int          i32x8 __attribute__((ext_vector_type(8)));

// ---- flexible 4-channel tensor accessors: split (3ch plane + 1ch plane) or
// ---- interleaved [B,4,H,W] -------------------------------------------------
__device__ __forceinline__ size_t o_off(int b, int c, int H, int W, int y, int x, int nch) {
  return (((size_t)b * nch + c) * H + y) * (size_t)W + x;
}
__device__ __forceinline__ float out_read(const float* p3, const float* p1, const float* pall,
                                          int interleaved, int b, int c, int H, int W, int y, int x) {
  if (interleaved) return pall[o_off(b, c, H, W, y, x, 4)];
  return (c < 3) ? p3[o_off(b, c, H, W, y, x, 3)]
                 : p1[((size_t)b * H + y) * (size_t)W + x];
}
__device__ __forceinline__ void out_write(float* p3, float* p1, float* pall,
                                          int interleaved, int b, int c, int H, int W, int y, int x, float v) {
  if (interleaved)        pall[o_off(b, c, H, W, y, x, 4)] = v;
  else if (c < 3)         p3[o_off(b, c, H, W, y, x, 3)] = v;
  else                    p1[((size_t)b * H + y) * (size_t)W + x] = v;
}

// monotonic float->uint key (ascending)
__device__ __forceinline__ unsigned fkey(float f) {
  unsigned u = __float_as_uint(f);
  return (u & 0x80000000u) ? ~u : (u | 0x80000000u);
}

// ---- 2x bilinear upsample, align_corners=True ------------------------------
__global__ void upsample2x_k(const float* i3, const float* i1, const float* iall, int iint,
                             float* o3, float* o1, float* oall, int oint,
                             int H, int W) {
  const int H2 = 2 * H, W2 = 2 * W;
  long long t = (long long)blockIdx.x * blockDim.x + threadIdx.x;
  long long total = (long long)NB * 4 * H2 * W2;
  if (t >= total) return;
  int x = (int)(t % W2); long long r = t / W2;
  int y = (int)(r % H2); r /= H2;
  int c = (int)(r % 4);  int b = (int)(r / 4);

  float sy = y * ((float)(H - 1) / (float)(H2 - 1));
  float sx = x * ((float)(W - 1) / (float)(W2 - 1));
  int y0 = (int)sy; int y1 = min(y0 + 1, H - 1); float wy = sy - (float)y0;
  int x0 = (int)sx; int x1 = min(x0 + 1, W - 1); float wx = sx - (float)x0;

  float v00 = out_read(i3, i1, iall, iint, b, c, H, W, y0, x0);
  float v01 = out_read(i3, i1, iall, iint, b, c, H, W, y0, x1);
  float v10 = out_read(i3, i1, iall, iint, b, c, H, W, y1, x0);
  float v11 = out_read(i3, i1, iall, iint, b, c, H, W, y1, x1);
  float v = (v00 * (1.f - wx) + v01 * wx) * (1.f - wy)
          + (v10 * (1.f - wx) + v11 * wx) * wy;
  out_write(o3, o1, oall, oint, b, c, H2, W2, y, x, v);
}

// ---- per-pixel uncertainty = second_max - max over 4 channels --------------
__global__ void uncert_k(const float* p3, const float* p1, const float* pall, int iint,
                         int H, int W, float* unc) {
  int b = blockIdx.y;
  int i = blockIdx.x * blockDim.x + threadIdx.x;
  int HW = H * W;
  if (i >= HW) return;
  int y = i / W, x = i % W;
  float a0 = out_read(p3, p1, pall, iint, b, 0, H, W, y, x);
  float a1 = out_read(p3, p1, pall, iint, b, 1, H, W, y, x);
  float a2 = out_read(p3, p1, pall, iint, b, 2, H, W, y, x);
  float a3 = out_read(p3, p1, pall, iint, b, 3, H, W, y, x);
  float lo01 = fminf(a0, a1), hi01 = fmaxf(a0, a1);
  float lo23 = fminf(a2, a3), hi23 = fmaxf(a2, a3);
  float m1 = fmaxf(hi01, hi23);
  float m2 = fmaxf(fminf(hi01, hi23), fmaxf(lo01, lo23));
  unc[(size_t)b * HW + i] = m2 - m1;   // == -(top1 - top2)
}

// ---- radix-select state: per batch 8 uints {prefix, mask, G, Nrem, ctrG, ctrT, -, -}
__global__ void init_k(unsigned* hist, unsigned* state, int N) {
  int t = blockIdx.x * blockDim.x + threadIdx.x;
  if (t < NB * 256) hist[t] = 0u;
  if (t < NB) {
    state[t * 8 + 0] = 0u; state[t * 8 + 1] = 0u;
    state[t * 8 + 2] = 0u; state[t * 8 + 3] = (unsigned)N;
    state[t * 8 + 4] = 0u; state[t * 8 + 5] = 0u;
  }
}

__global__ void hist_k(const float* unc, int HW, const unsigned* state, unsigned* hist, int shift) {
  int b = blockIdx.y;
  __shared__ unsigned lh[256];
  if (threadIdx.x < 256) lh[threadIdx.x] = 0u;
  __syncthreads();
  unsigned prefix = state[b * 8 + 0];
  unsigned mask   = state[b * 8 + 1];
  int stride = gridDim.x * blockDim.x;
  const float* up = unc + (size_t)b * HW;
  for (int i = blockIdx.x * blockDim.x + threadIdx.x; i < HW; i += stride) {
    __builtin_prefetch(up + i + stride, 0, 0);   // gfx1250 global_prefetch_b8
    unsigned u = fkey(up[i]);
    if ((u & mask) == prefix) atomicAdd(&lh[(u >> shift) & 0xFFu], 1u);
  }
  __syncthreads();
  if (threadIdx.x < 256 && lh[threadIdx.x]) atomicAdd(&hist[b * 256 + threadIdx.x], lh[threadIdx.x]);
}

__global__ void pick_k(unsigned* hist, unsigned* state, int shift) {
  int b = blockIdx.x;
  __shared__ unsigned h[256];
  h[threadIdx.x] = hist[b * 256 + threadIdx.x];
  hist[b * 256 + threadIdx.x] = 0u;             // reset for next pass
  __syncthreads();
  if (threadIdx.x == 0) {
    unsigned Nrem = state[b * 8 + 3];
    unsigned cum = 0; int bin = 255;
    for (; bin > 0; --bin) { cum += h[bin]; if (cum >= Nrem) break; }
    if (cum < Nrem) cum += h[0];                // bin==0 fallthrough
    unsigned above = cum - h[bin];
    state[b * 8 + 0] |= ((unsigned)bin) << shift;
    state[b * 8 + 1] |= 0xFFu << shift;
    state[b * 8 + 2] += above;
    state[b * 8 + 3] = Nrem - above;
  }
}

__global__ void compact_k(const float* unc, int HW, unsigned* state, int* idxsel, int N) {
  int b = blockIdx.y;
  int i = blockIdx.x * blockDim.x + threadIdx.x;
  if (i >= HW) return;
  unsigned T    = state[b * 8 + 0];
  unsigned G    = state[b * 8 + 2];
  unsigned ties = state[b * 8 + 3];
  unsigned u = fkey(unc[(size_t)b * HW + i]);
  if (u > T) {
    unsigned p = atomicAdd(&state[b * 8 + 4], 1u);
    idxsel[(size_t)b * N + p] = i;
  } else if (u == T) {
    unsigned t = atomicAdd(&state[b * 8 + 5], 1u);
    if (t < ties) idxsel[(size_t)b * N + G + t] = i;
  }
}

// ---- fused point-sample + WMMA MLP + scatter -------------------------------
// One wave = 16 points. A (16x32 f16) = zero-padded mlp_w; B (32x16 f16) =
// feature tile [4 coarse + 16 hid + 12 zero] per point. D rows 0..3 scattered.
// Weights are staged into LDS via the Tensor Data Mover (TDM).
__global__ void refine_k(const int* idxsel, int N,
                         const float* hid,
                         float* o3, float* o1, float* oall, int oint,
                         const float* mlp_w, const float* mlp_b,
                         int H, int W) {
  __shared__ float smem[80 + 16 * 32 + 4];  // raw W | padded 16x32 W | bias
  float* sWraw = smem;        // LDS offset 0 (first/only shared allocation)
  float* sW    = smem + 80;
  float* sb    = smem + 80 + 16 * 32;

#if __has_builtin(__builtin_amdgcn_tensor_load_to_lds) && \
    __has_builtin(__builtin_amdgcn_s_wait_tensorcnt)
  {
    // ---- D# per CDNA5 ISA ch.8: 1-row tile of 80 f32 (320B), global->LDS ----
    unsigned long long ga = (unsigned long long)(size_t)mlp_w;
    u32x4 g0;
    g0[0] = 1u;                                  // count=1 valid descriptor
    g0[1] = 0u;                                  // lds_addr = 0 (start of smem)
    g0[2] = (unsigned)(ga & 0xFFFFFFFFull);      // global_addr[31:0]
    g0[3] = (unsigned)((ga >> 32) & 0x01FFFFFFull) | (2u << 30);  // ga[56:32] | type=2
    i32x8 g1;
    g1[0] = 0x00020000;      // data_size=4B (bits 17:16 = 2), wg_mask=0
    g1[1] = 80 << 16;        // tensor_dim0[15:0] in bits 63:48
    g1[2] = 1 << 16;         // tensor_dim0[31:16]=0 | tensor_dim1[15:0]=1 in bits 95:80
    g1[3] = 80 << 16;        // tensor_dim1[31:16]=0 | tile_dim0=80 in bits 127:112
    g1[4] = 1;               // tile_dim1=1, tile_dim2=0
    g1[5] = 80;              // tensor_dim0_stride[31:0]
    g1[6] = 0;               // tensor_dim0_stride[47:32]=0, tensor_dim1_stride lo=0
    g1[7] = 0;
    i32x4 g2 = {0, 0, 0, 0};
    i32x4 g3 = {0, 0, 0, 0};
#if defined(__clang_major__) && (__clang_major__ >= 23)
    i32x8 g4 = {0, 0, 0, 0, 0, 0, 0, 0};
    __builtin_amdgcn_tensor_load_to_lds(g0, g1, g2, g3, g4, 0);
#else
    __builtin_amdgcn_tensor_load_to_lds(g0, g1, g2, g3, 0);
#endif
    __builtin_amdgcn_s_wait_tensorcnt(0);        // s_wait_tensorcnt 0x0
  }
#else
  for (int i = threadIdx.x; i < 80; i += blockDim.x) sWraw[i] = mlp_w[i];
#endif
  __syncthreads();

  // expand raw 4x20 weights into zero-padded 16x32 A-matrix storage
  for (int i = threadIdx.x; i < 16 * 32; i += blockDim.x) {
    int r = i >> 5, k = i & 31;
    sW[i] = (r < 4 && k < 20) ? sWraw[r * 20 + k] : 0.f;
  }
  if (threadIdx.x < 4) sb[threadIdx.x] = mlp_b[threadIdx.x];
  __syncthreads();

  const int lane = threadIdx.x & 31;
  const int wave = threadIdx.x >> 5;
  const int b    = blockIdx.y;
  const int tile = blockIdx.x * (blockDim.x >> 5) + wave;
  const int col  = lane & 15;          // matrix column == point within tile
  const int hi   = lane >> 4;          // 0: K{0..7,16..23}, 1: K{8..15,24..31}
  const int n    = tile * 16 + col;

  const int idx = idxsel[(size_t)b * N + n];
  const int px = idx % W, py = idx / W;

  // ---- A fragment (ISA 16-bit 16x32 layout: lanes<16 -> K{0..7,16..23},
  // ---- lanes>=16 -> K{8..15,24..31}; row M = lane&15) ----------------------
  v16h a;
  {
    const int row = col;
    const int kA = hi * 8, kB = 16 + hi * 8;
#pragma unroll
    for (int j = 0; j < 8; ++j) a[j]     = (_Float16)sW[row * 32 + kA + j];
#pragma unroll
    for (int j = 0; j < 8; ++j) a[8 + j] = (_Float16)sW[row * 32 + kB + j];
  }

  // ---- bilinear setup for hid (align_corners=False grid_sample) ------------
  float xf = ((px + 0.5f) / (float)W) * (float)HSZ - 0.5f;
  float yf = ((py + 0.5f) / (float)H) * (float)HSZ - 0.5f;
  float x0f = floorf(xf), y0f = floorf(yf);
  float wx = xf - x0f, wy = yf - y0f;
  int x0 = min(max((int)x0f, 0), HSZ - 1); int x1 = min(x0 + 1, HSZ - 1);
  int y0 = min(max((int)y0f, 0), HSZ - 1); int y1 = min(y0 + 1, HSZ - 1);
  float w00 = (1.f - wy) * (1.f - wx), w01 = (1.f - wy) * wx;
  float w10 = wy * (1.f - wx),         w11 = wy * wx;

  // ---- B fragment (ISA 16-bit 32x16: lane<16 -> K{0..7,16..23} of col=lane,
  // ---- lane>=16 -> K{8..15,24..31} of col=lane-16) -------------------------
  v16h bf;
  const int kbase = hi * 16;
#pragma unroll
  for (int j = 0; j < 16; ++j) {
    int k = ((j < 8) ? 0 : 8) + hi * 8 + j;
    float v = 0.f;
    if (k < 4) {
      // coarse sample degenerates to exact pixel gather at own resolution
      v = out_read(o3, o1, oall, oint, b, k, H, W, py, px);
    } else if (k < 20) {
      int hc = k - 4;
      const float* hp = hid + ((size_t)b * 16 + hc) * (size_t)HSZ * HSZ;
      v = w00 * hp[(size_t)y0 * HSZ + x0] + w01 * hp[(size_t)y0 * HSZ + x1]
        + w10 * hp[(size_t)y1 * HSZ + x0] + w11 * hp[(size_t)y1 * HSZ + x1];
    }
    bf[j] = (_Float16)v;
  }

  v8f c = {};
  c = __builtin_amdgcn_wmma_f32_16x16x32_f16(
      /*neg_a=*/false, a, /*neg_b=*/false, bf,
      /*c_mod=*/(short)0, c, /*reuse_a=*/false, /*reuse_b=*/false);

  // D layout: lanes 0..15, VGPR r -> D[M=r][N=lane]; we need rows 0..3
  if (hi == 0) {
#pragma unroll
    for (int r = 0; r < 4; ++r) {
      out_write(o3, o1, oall, oint, b, r, H, W, py, px, c[r] + sb[r]);
    }
  }
}

// ---------------------------------------------------------------------------
extern "C" void kernel_launch(void* const* d_in, const int* in_sizes, int n_in,
                              void* d_out, int out_size, void* d_ws, size_t ws_size,
                              hipStream_t stream) {
  (void)in_sizes; (void)n_in; (void)out_size; (void)ws_size;
  const float* hid   = (const float*)d_in[1];   // [8,16,512,512]
  const float* fgr   = (const float*)d_in[2];   // [8,3,128,128]
  const float* pha   = (const float*)d_in[3];   // [8,1,128,128]
  const float* mlp_w = (const float*)d_in[4];   // [4,20]
  const float* mlp_b = (const float*)d_in[5];   // [4]

  float* outF = (float*)d_out;                          // [8,3,512,512]
  float* outP = outF + (size_t)NB * 3 * HSZ * HSZ;      // [8,1,512,512]

  // workspace carve-up (~17 MB)
  float*    out256 = (float*)d_ws;                              // 8*4*256*256
  float*    unc    = out256 + (size_t)NB * 4 * 256 * 256;       // 8*512*512
  int*      idxsel = (int*)(unc + (size_t)NB * HSZ * HSZ);      // 8*4096
  unsigned* hist   = (unsigned*)(idxsel + NB * 4096);           // 8*256
  unsigned* state  = hist + NB * 256;                           // 8*8

  // ================= iteration 1: 128 -> 256 =================
  {
    const int H = 128, W = 128, H2 = 256, W2 = 256, HW = H2 * W2, N = 1024;
    upsample2x_k<<<(NB * 4 * HW + 255) / 256, 256, 0, stream>>>(
        fgr, pha, nullptr, 0, nullptr, nullptr, out256, 1, H, W);
    uncert_k<<<dim3(HW / 256, NB), 256, 0, stream>>>(
        nullptr, nullptr, out256, 1, H2, W2, unc);
    init_k<<<NB, 256, 0, stream>>>(hist, state, N);
    for (int p = 0; p < 4; ++p) {
      int sh = 24 - 8 * p;
      hist_k<<<dim3(128, NB), 256, 0, stream>>>(unc, HW, state, hist, sh);
      pick_k<<<NB, 256, 0, stream>>>(hist, state, sh);
    }
    compact_k<<<dim3(HW / 256, NB), 256, 0, stream>>>(unc, HW, state, idxsel, N);
    refine_k<<<dim3(N / 128, NB), 256, 0, stream>>>(
        idxsel, N, hid, nullptr, nullptr, out256, 1, mlp_w, mlp_b, H2, W2);
  }

  // ================= iteration 2: 256 -> 512 (into d_out, split) =============
  {
    const int H = 256, W = 256, H2 = 512, W2 = 512, HW = H2 * W2, N = 4096;
    upsample2x_k<<<(NB * 4 * HW + 255) / 256, 256, 0, stream>>>(
        nullptr, nullptr, out256, 1, outF, outP, nullptr, 0, H, W);
    uncert_k<<<dim3(HW / 256, NB), 256, 0, stream>>>(
        outF, outP, nullptr, 0, H2, W2, unc);
    init_k<<<NB, 256, 0, stream>>>(hist, state, N);
    for (int p = 0; p < 4; ++p) {
      int sh = 24 - 8 * p;
      hist_k<<<dim3(128, NB), 256, 0, stream>>>(unc, HW, state, hist, sh);
      pick_k<<<NB, 256, 0, stream>>>(hist, state, sh);
    }
    compact_k<<<dim3(HW / 256, NB), 256, 0, stream>>>(unc, HW, state, idxsel, N);
    refine_k<<<dim3(N / 128, NB), 256, 0, stream>>>(
        idxsel, N, hid, outF, outP, nullptr, 0, mlp_w, mlp_b, H2, W2);
  }
}
